// FasterGRUTree_53163105190066
// MI455X (gfx1250) — matrix operand order repaired and checked by reading, not compiled, not run on hardware
//
#include <hip/hip_runtime.h>
#include <hip/hip_bf16.h>

// ---- problem constants ----
#define MEMD   150
#define NPAD   480     // 3 gates * 160 (per-gate padded to 160)
#define NTILES 30      // 480 / 16
#define GPAD   160
#define KCH    256     // ch input K (250) padded
#define KTCH   8
#define KHPAD  160     // hidden K (150) padded
#define KTH    5
#define KLEAF  352     // leaf input K (350) padded
#define KTLEAF 11
#define ASTR   416     // level sA row stride (KCH + KHPAD) bf16 elems
#define HOFF   256     // hidden region offset inside sA row
#define HSTR   160     // h ping-pong buffer row stride (bf16 elems)
#define NNODES 1023
#define BATCH  64
#define LEAVES 512

typedef __bf16 v16bf __attribute__((ext_vector_type(16)));
typedef float  v8f   __attribute__((ext_vector_type(8)));

union ABFrag { uint4 q[2]; v16bf v; };

__device__ __forceinline__ unsigned short f2bf(float f) {
  unsigned int u = __float_as_uint(f);
  unsigned int r = u + 0x7FFFu + ((u >> 16) & 1u);   // round-to-nearest-even
  return (unsigned short)(r >> 16);
}
__device__ __forceinline__ float bf2f(unsigned short h) {
  return __uint_as_float(((unsigned int)h) << 16);
}
__device__ __forceinline__ float sigm(float x) { return 1.0f / (1.0f + __expf(-x)); }

// One 16x16 output tile, K-loop over pre-swizzled bf16 fragments.
// A: LDS bf16, ISA 16-bit A layout assembled with two b128 LDS reads per k-chunk.
// B: global pre-packed fragments, layout dword idx = ((nt*KT+kt)*32 + lane)*8 + i.
__device__ __forceinline__ v8f gemm_tile(const unsigned short* __restrict__ sA, int astride, int aoff,
                                         const unsigned int* __restrict__ frag, int nt, int KT,
                                         int lane, v8f c)
{
  const int m  = lane & 15;
  const int g8 = (lane >> 4) << 3;                     // 0 or 8
  const unsigned short* arow = sA + m * astride + aoff + g8;
  const uint4* bptr = ((const uint4*)frag) + ((size_t)(nt * KT) * 32 + lane) * 2;
  for (int kt = 0; kt < KT; ++kt) {
    ABFrag a, b;
    a.q[0] = *(const uint4*)(arow + kt * 32);          // K = base .. base+7
    a.q[1] = *(const uint4*)(arow + kt * 32 + 16);     // K = base+16 .. base+23
    b.q[0] = bptr[0];
    b.q[1] = bptr[1];
    bptr += 64;                                        // next k-chunk (32 lanes * 2 uint4)
    c = __builtin_amdgcn_wmma_f32_16x16x32_bf16(false, a.v, false, b.v, (short)0, c, false, false);
  }
  return c;
}

// C layout: VGPR v, lanes 0-15 -> row v, lanes 16-31 -> row v+8; col = lane&15.
__device__ __forceinline__ void store_tile(float* __restrict__ dst, int stride, int colbase,
                                           int lane, v8f c)
{
  const int col = colbase + (lane & 15);
  const int r0  = (lane >> 4) << 3;
#pragma unroll
  for (int v = 0; v < 8; ++v) dst[(size_t)(r0 + v) * stride + col] = c[v];
}

// Pack an f32 weight (450 x Kreal, row-major) into bf16 B-fragments with
// per-gate N padding (gate g, d<150 -> logical col g*160+d) and K zero-padding.
__global__ __launch_bounds__(256) void pack_kernel(const float* __restrict__ W,
                                                   unsigned int* __restrict__ dst,
                                                   int Kreal, int KT)
{
  int t = blockIdx.x * 256 + threadIdx.x;
  int total = NTILES * KT * 256;
  if (t >= total) return;
  int i    = t & 7;
  int lane = (t >> 3) & 31;
  int fidx = t >> 8;            // nt*KT + kt
  int kt = fidx % KT;
  int nt = fidx / KT;
  int nlog = nt * 16 + (lane & 15);
  int g = nlog / 160, d = nlog % 160;
  int nrow = g * 150 + d;
  int k0 = kt * 32 + ((lane >> 4) << 3) + 2 * i + ((i >> 2) << 3);   // lo lanes K 0..7,16..23; hi lanes K 8..15,24..31
  float v0 = 0.f, v1 = 0.f;
  if (d < 150) {
    if (k0     < Kreal) v0 = W[(size_t)nrow * Kreal + k0];
    if (k0 + 1 < Kreal) v1 = W[(size_t)nrow * Kreal + k0 + 1];
  }
  dst[t] = (unsigned int)f2bf(v0) | ((unsigned int)f2bf(v1) << 16);
}

// Leaf GRU: h = (1-z)*n with gh = bhh (h0 == 0). One 16-row block per workgroup.
__global__ __launch_bounds__(256) void leaf_kernel(const float* __restrict__ embs,
                                                   const float* __restrict__ tags,
                                                   const unsigned int* __restrict__ fragW,
                                                   const float* __restrict__ bih,
                                                   const float* __restrict__ bhh,
                                                   unsigned short* __restrict__ hout)
{
  __shared__ unsigned short sA[16 * KLEAF];
  __shared__ float sGI[16 * NPAD];
  const int tid = threadIdx.x;
  const int wave = tid >> 5, lane = tid & 31;
  const int rbase = blockIdx.x * 16;

  for (int e = tid; e < 16 * KLEAF; e += 256) {
    int m = e / KLEAF, k = e % KLEAF;
    int r = rbase + m, b = r >> 9, leaf = r & (LEAVES - 1);
    size_t node = (size_t)b * NNODES + leaf;
    float v = 0.f;
    if (k < 300)      v = embs[node * 300 + k];
    else if (k < 350) v = tags[node * 50 + (k - 300)];
    sA[m * KLEAF + k] = f2bf(v);
  }
  __syncthreads();

  for (int j = 0; j < 4; ++j) {
    int nt = wave + 8 * j;
    if (nt >= NTILES) break;                            // wave-uniform
    v8f c = {};
    c = gemm_tile(sA, KLEAF, 0, fragW, nt, KTLEAF, lane, c);
    store_tile(sGI, NPAD, nt * 16, lane, c);
  }
  __syncthreads();

  for (int e = tid; e < 16 * MEMD; e += 256) {
    int m = e / MEMD, d = e % MEMD;
    const float* gi = sGI + m * NPAD;
    float r = sigm(gi[d]       + bih[d]       + bhh[d]);
    float z = sigm(gi[160 + d] + bih[150 + d] + bhh[150 + d]);
    float n = tanhf(gi[320 + d] + bih[300 + d] + r * bhh[300 + d]);
    float h = (1.f - z) * n;
    int rr = rbase + m;
    hout[(size_t)rr * HSTR + d] = f2bf(h);
  }
}

// One tree level: ch-GRU(left, h=0) -> ch-GRU(right, hg1) -> node-GRU(x=0, hg2).
__global__ __launch_bounds__(256) void level_kernel(const unsigned short* __restrict__ hin,
                                                    const float* __restrict__ tags,
                                                    const unsigned int* __restrict__ fragChW,
                                                    const unsigned int* __restrict__ fragChU,
                                                    const unsigned int* __restrict__ fragNodeU,
                                                    const float* __restrict__ chbih,
                                                    const float* __restrict__ chbhh,
                                                    const float* __restrict__ nbih,
                                                    const float* __restrict__ nbhh,
                                                    unsigned short* __restrict__ hout,
                                                    float* __restrict__ outf,
                                                    int cnt, int offset)
{
  __shared__ unsigned short sA[16 * ASTR];   // [0,KCH): concat input; [HOFF,HOFF+KHPAD): hg bf16
  __shared__ float sGI[16 * NPAD];
  __shared__ float sGN[16 * GPAD];
  const int tid = threadIdx.x;
  const int wave = tid >> 5, lane = tid & 31;
  const int rbase = blockIdx.x * 16;
  const int pcnt = cnt >> 1;
  const int poff = offset + cnt;

  auto stage = [&](int right) {
    for (int e = tid; e < 16 * KCH; e += 256) {
      int m = e / KCH, k = e % KCH;
      int r = rbase + m, b = r / pcnt, p = r % pcnt;
      int child = 2 * p + right;
      float v = 0.f;
      if (k < MEMD)     v = bf2f(hin[(size_t)(b * cnt + child) * HSTR + k]);
      else if (k < 200) v = tags[((size_t)b * NNODES + offset + child) * 50 + (k - 150)];
      else if (k < 250) v = tags[((size_t)b * NNODES + poff + p) * 50 + (k - 200)];
      sA[m * ASTR + k] = f2bf(v);
    }
  };

  // ---- stage A1 (left child) + zero hidden-region K padding ----
  stage(0);
  for (int e = tid; e < 16 * (KHPAD - MEMD); e += 256) {
    int m = e / (KHPAD - MEMD), d = MEMD + e % (KHPAD - MEMD);
    sA[m * ASTR + HOFF + d] = 0;
  }
  __syncthreads();

  // ---- GEMM gi1 ----
  for (int j = 0; j < 4; ++j) {
    int nt = wave + 8 * j;
    if (nt >= NTILES) break;
    v8f c = {};
    c = gemm_tile(sA, ASTR, 0, fragChW, nt, KTCH, lane, c);
    store_tile(sGI, NPAD, nt * 16, lane, c);
  }
  __syncthreads();

  // ---- cell 1 elementwise: hg1 (h_prev = 0, gh = bhh) -> sA hidden region ----
  for (int e = tid; e < 16 * MEMD; e += 256) {
    int m = e / MEMD, d = e % MEMD;
    const float* gi = sGI + m * NPAD;
    float r = sigm(gi[d]       + chbih[d]       + chbhh[d]);
    float z = sigm(gi[160 + d] + chbih[150 + d] + chbhh[150 + d]);
    float n = tanhf(gi[320 + d] + chbih[300 + d] + r * chbhh[300 + d]);
    sA[m * ASTR + HOFF + d] = f2bf((1.f - z) * n);
  }
  __syncthreads();

  // ---- stage A2 (right child) ----
  stage(1);
  __syncthreads();

  // ---- fused GEMM: gi2 (+ gh2 accumulated for r/z tiles; n-gate gh2 -> sGN) ----
  for (int j = 0; j < 4; ++j) {
    int nt = wave + 8 * j;
    if (nt >= NTILES) break;
    v8f c = {};
    c = gemm_tile(sA, ASTR, 0, fragChW, nt, KTCH, lane, c);
    if (nt < 20) {                                      // r,z gate tiles: i + h summed in regs
      c = gemm_tile(sA, ASTR, HOFF, fragChU, nt, KTH, lane, c);
      store_tile(sGI, NPAD, nt * 16, lane, c);
    } else {                                            // n gate: keep i_n and h_n separate
      store_tile(sGI, NPAD, nt * 16, lane, c);
      v8f c2 = {};
      c2 = gemm_tile(sA, ASTR, HOFF, fragChU, nt, KTH, lane, c2);
      store_tile(sGN, GPAD, (nt - 20) * 16, lane, c2);
    }
  }
  __syncthreads();

  // ---- cell 2 elementwise: hg2 (h_prev = hg1) ----
  for (int e = tid; e < 16 * MEMD; e += 256) {
    int m = e / MEMD, d = e % MEMD;
    const float* gi = sGI + m * NPAD;
    float r  = sigm(gi[d]       + chbih[d]       + chbhh[d]);
    float z  = sigm(gi[160 + d] + chbih[150 + d] + chbhh[150 + d]);
    float hn = sGN[m * GPAD + d] + chbhh[300 + d];
    float n  = tanhf(gi[320 + d] + chbih[300 + d] + r * hn);
    float hp = bf2f(sA[m * ASTR + HOFF + d]);
    sA[m * ASTR + HOFF + d] = f2bf((1.f - z) * n + z * hp);
  }
  __syncthreads();

  // ---- GEMM gh3 = hg2 @ node_Whh^T ----
  for (int j = 0; j < 4; ++j) {
    int nt = wave + 8 * j;
    if (nt >= NTILES) break;
    v8f c = {};
    c = gemm_tile(sA, ASTR, HOFF, fragNodeU, nt, KTH, lane, c);
    store_tile(sGI, NPAD, nt * 16, lane, c);
  }
  __syncthreads();

  // ---- node elementwise: gi = node_bih (x = 0), h_prev = hg2 ----
  for (int e = tid; e < 16 * MEMD; e += 256) {
    int m = e / MEMD, d = e % MEMD;
    const float* gh = sGI + m * NPAD;
    float r = sigm(nbih[d]       + gh[d]       + nbhh[d]);
    float z = sigm(nbih[150 + d] + gh[160 + d] + nbhh[150 + d]);
    float n = tanhf(nbih[300 + d] + r * (gh[320 + d] + nbhh[300 + d]));
    float hp = bf2f(sA[m * ASTR + HOFF + d]);
    float h = (1.f - z) * n + z * hp;
    int rr = rbase + m;
    if (outf) outf[(size_t)rr * MEMD + d] = h;
    else      hout[(size_t)rr * HSTR + d] = f2bf(h);
  }
}

extern "C" void kernel_launch(void* const* d_in, const int* in_sizes, int n_in,
                              void* d_out, int out_size, void* d_ws, size_t ws_size,
                              hipStream_t stream)
{
  (void)in_sizes; (void)n_in; (void)out_size; (void)ws_size;
  const float* embs     = (const float*)d_in[0];
  const float* tags     = (const float*)d_in[1];
  const float* leaf_Wih = (const float*)d_in[2];
  // d_in[3] leaf_Whh unused (h0 == 0)
  const float* leaf_bih = (const float*)d_in[4];
  const float* leaf_bhh = (const float*)d_in[5];
  // d_in[6] node_Wih unused (x == 0)
  const float* node_Whh = (const float*)d_in[7];
  const float* node_bih = (const float*)d_in[8];
  const float* node_bhh = (const float*)d_in[9];
  const float* ch_Wih   = (const float*)d_in[10];
  const float* ch_Whh   = (const float*)d_in[11];
  const float* ch_bih   = (const float*)d_in[12];
  const float* ch_bhh   = (const float*)d_in[13];

  // workspace layout
  unsigned int* fragLeaf  = (unsigned int*)d_ws;
  unsigned int* fragChW   = fragLeaf  + (size_t)NTILES * KTLEAF * 256;
  unsigned int* fragChU   = fragChW   + (size_t)NTILES * KTCH   * 256;
  unsigned int* fragNodeU = fragChU   + (size_t)NTILES * KTH    * 256;
  unsigned short* hbuf0   = (unsigned short*)(fragNodeU + (size_t)NTILES * KTH * 256);
  unsigned short* hbuf1   = hbuf0 + (size_t)BATCH * LEAVES * HSTR;

  // pack weights into WMMA B-fragments (bf16, pre-swizzled)
  {
    int t;
    t = NTILES * KTLEAF * 256;
    pack_kernel<<<(t + 255) / 256, 256, 0, stream>>>(leaf_Wih, fragLeaf, 350, KTLEAF);
    t = NTILES * KTCH * 256;
    pack_kernel<<<(t + 255) / 256, 256, 0, stream>>>(ch_Wih, fragChW, 250, KTCH);
    t = NTILES * KTH * 256;
    pack_kernel<<<(t + 255) / 256, 256, 0, stream>>>(ch_Whh, fragChU, 150, KTH);
    pack_kernel<<<(t + 255) / 256, 256, 0, stream>>>(node_Whh, fragNodeU, 150, KTH);
  }

  // leaf level: 64*512 rows, 16 rows per workgroup
  leaf_kernel<<<(BATCH * LEAVES) / 16, 256, 0, stream>>>(embs, tags, fragLeaf,
                                                         leaf_bih, leaf_bhh, hbuf0);

  // tree levels
  unsigned short* hin = hbuf0;
  unsigned short* hot = hbuf1;
  int offset = 0, cnt = LEAVES;
  while (cnt > 1) {
    int pcnt = cnt >> 1;
    float* outf = (pcnt == 1) ? (float*)d_out : nullptr;
    level_kernel<<<(BATCH * pcnt) / 16, 256, 0, stream>>>(hin, tags, fragChW, fragChU, fragNodeU,
                                                          ch_bih, ch_bhh, node_bih, node_bhh,
                                                          hot, outf, cnt, offset);
    unsigned short* t = hin; hin = hot; hot = t;
    offset += cnt; cnt = pcnt;
  }
}
